// EdgeMessagePassingLayer_180388626631
// MI455X (gfx1250) — matrix operand order repaired and checked by reading, not compiled, not run on hardware
//
#include <hip/hip_runtime.h>

#define H_DIM 128
#define K_MSG 256          // H + ED + GD
#define K_UPD 320          // 2H + GD
#define K_PAD (K_MSG + 8)  // 264: LDS row stride, bank-conflict-free, 16B aligned
#define H_PAD (H_DIM + 8)  // 136
#define KU_PAD (K_UPD + 8) // 328

#if defined(__has_builtin)
#if __has_builtin(__builtin_amdgcn_sched_barrier)
#define SCHED_FENCE() __builtin_amdgcn_sched_barrier(0)
#endif
#endif
#ifndef SCHED_FENCE
#define SCHED_FENCE()
#endif

typedef __attribute__((ext_vector_type(16))) __bf16 bf16x16;
typedef __attribute__((ext_vector_type(8)))  float  f32x8;

union FragU { bf16x16 v; uint4 q[2]; };

// ---- bf16 conversion (round-to-nearest-even) ----
__device__ __forceinline__ unsigned pk2(float a, float b) {
  unsigned ua = __builtin_bit_cast(unsigned, a);
  unsigned ub = __builtin_bit_cast(unsigned, b);
  ua = (ua + 0x7FFFu + ((ua >> 16) & 1u)) >> 16;
  ub = (ub + 0x7FFFu + ((ub >> 16) & 1u)) >> 16;
  return ua | (ub << 16);
}
__device__ __forceinline__ __bf16 tobf(float x) {
  unsigned u = __builtin_bit_cast(unsigned, x);
  u = (u + 0x7FFFu + ((u >> 16) & 1u)) >> 16;
  unsigned short s = (unsigned short)u;
  return __builtin_bit_cast(__bf16, s);
}

// ---- WMMA fragment loaders ----
// A (16x32): lanes 0-15 hold M=lane, K {0..7,16..23}; lanes 16-31 M=lane-16,
// K {8..15,24..31}. base = row start + k_tile*32 elements.
__device__ __forceinline__ bf16x16 load_a_frag(const __bf16* base, int half) {
  FragU f;
  f.q[0] = *(const uint4*)(base + 8 * half);
  f.q[1] = *(const uint4*)(base + 16 + 8 * half);
  return f.v;
}
// B (32x16): lanes 0-15 hold K {0..7,16..23} of column N=lane; lanes 16-31 hold
// K {8..15,24..31} of column N=lane-16. base = column row start + k_tile*32 elements.
__device__ __forceinline__ bf16x16 load_b_frag(const __bf16* base, int half) {
  FragU f;
  const __bf16* p = base + 8 * half;
  f.q[0] = *(const uint4*)(p);
  f.q[1] = *(const uint4*)(p + 16);
  return f.v;
}
__device__ __forceinline__ f32x8 wmma_bf16(bf16x16 a, bf16x16 b, f32x8 c) {
  return __builtin_amdgcn_wmma_f32_16x16x32_bf16(false, a, false, b, (short)0, c,
                                                 false, false);
}

// ---- weight repack: fp32 [K][N] row-major -> bf16 [N][K] ----
__global__ void prep_kernel(const float* __restrict__ w, __bf16* __restrict__ wt,
                            int K, int Nn) {
  int i = blockIdx.x * blockDim.x + threadIdx.x;
  if (i < K * Nn) {
    int k = i / Nn, n = i - k * Nn;
    wt[n * K + k] = tobf(w[i]);
  }
}

// ================= message MLP + scatter aggregation =================
// A = weight tile (16 outputs x K) persistent in VGPRs; B = staged input tile
// (K x 16 edges) from LDS. D(16x16): M = output dim, N = edge. 8 waves cover
// the 128 outputs. B fragments are preloaded and a sched_barrier pins the
// loads above the WMMA chain so LDS latency is paid once per GEMM.
__global__ void __launch_bounds__(256) msg_kernel(
    const float* __restrict__ node, const long long* __restrict__ eidx,
    const float* __restrict__ est, const float* __restrict__ gst,
    const float* __restrict__ mb1, const float* __restrict__ mb2,
    const __bf16* __restrict__ W1t, const __bf16* __restrict__ W2t,
    float* __restrict__ agg, float* __restrict__ deg, int E) {
  __shared__ __bf16 sIn[16 * K_PAD];  // [edge][K] input tile, padded rows
  __shared__ __bf16 sHd[16 * H_PAD];  // [edge][H] hidden tile, padded rows
  __shared__ unsigned sDst[16];

  const int w    = threadIdx.x >> 5;  // wave = output tile (0..7)
  const int lane = threadIdx.x & 31;
  const int half = lane >> 4;
  const int l15  = lane & 15;
  const f32x8 z8 = {0.f, 0.f, 0.f, 0.f, 0.f, 0.f, 0.f, 0.f};

  // Persistent weight A-fragments (one-time global load, L2 hit).
  bf16x16 w1f[8], w2f[4];
#pragma unroll
  for (int kt = 0; kt < 8; kt++)
    w1f[kt] = load_a_frag(W1t + (w * 16 + l15) * K_MSG + kt * 32, half);
#pragma unroll
  for (int kt = 0; kt < 4; kt++)
    w2f[kt] = load_a_frag(W2t + (w * 16 + l15) * H_DIM + kt * 32, half);
  // Bias values for this lane's 8 output rows.
  float b1v[8], b2v[8];
#pragma unroll
  for (int r = 0; r < 8; r++) {
    b1v[r] = mb1[w * 16 + r + 8 * half];
    b2v[r] = mb2[w * 16 + r + 8 * half];
  }

  const long long tiles = ((long long)E + 15) >> 4;
  for (long long t = blockIdx.x; t < tiles; t += gridDim.x) {
    const long long tbase = t * 16;

    {  // stage gathered inputs -> bf16 LDS tile; 256 threads, 16 cols each
      const int row = threadIdx.x >> 4;       // edge in tile
      const int cg  = threadIdx.x & 15;       // column group (16 cols)
      const long long e = tbase + row;
      unsigned* dp = (unsigned*)(sIn + row * K_PAD + cg * 16);
      if (e < E) {
        const float4* p4;
        if (cg < 8) {
          const long long s = eidx[e];
          p4 = (const float4*)(node + s * (long long)H_DIM + cg * 16);
        } else if (cg < 12) {
          p4 = (const float4*)(est + e * 64 + (cg - 8) * 16);
        } else {
          p4 = (const float4*)(gst + (cg - 12) * 16);
        }
#pragma unroll
        for (int i = 0; i < 4; i++) {
          float4 v = p4[i];
          dp[2 * i] = pk2(v.x, v.y);
          dp[2 * i + 1] = pk2(v.z, v.w);
        }
      } else {
#pragma unroll
        for (int i = 0; i < 8; i++) dp[i] = 0u;
      }
      if (threadIdx.x < 16) {
        const long long e2 = tbase + threadIdx.x;
        sDst[threadIdx.x] = (e2 < E) ? (unsigned)eidx[(long long)E + e2] : 0u;
      }
    }
    __syncthreads();

    // GEMM1: W1[16xK] @ In^T[Kx16] — preload all B fragments, fence, then MMA
    {
      bf16x16 bfr[8];
#pragma unroll
      for (int kt = 0; kt < 8; kt++)
        bfr[kt] = load_b_frag(sIn + l15 * K_PAD + kt * 32, half);
      SCHED_FENCE();  // keep all 16 ds_load_b128 above the WMMA chain
      f32x8 acc = z8;
#pragma unroll
      for (int kt = 0; kt < 8; kt++) acc = wmma_bf16(w1f[kt], bfr[kt], acc);
      // bias + ReLU -> hidden: lane's 8 outputs are contiguous -> 1 b128 store
      unsigned hp[4];
#pragma unroll
      for (int r2 = 0; r2 < 4; r2++) {
        float v0 = fmaxf(acc[2 * r2] + b1v[2 * r2], 0.f);
        float v1 = fmaxf(acc[2 * r2 + 1] + b1v[2 * r2 + 1], 0.f);
        hp[r2] = pk2(v0, v1);
      }
      uint4 hq; hq.x = hp[0]; hq.y = hp[1]; hq.z = hp[2]; hq.w = hp[3];
      *(uint4*)(sHd + l15 * H_PAD + w * 16 + 8 * half) = hq;
    }
    __syncthreads();

    // GEMM2: W2[16x128] @ Hd^T[128x16] — preload, fence, MMA
    {
      bf16x16 bfr[4];
#pragma unroll
      for (int kt = 0; kt < 4; kt++)
        bfr[kt] = load_b_frag(sHd + l15 * H_PAD + kt * 32, half);
      SCHED_FENCE();
      f32x8 acc2 = z8;
#pragma unroll
      for (int kt = 0; kt < 4; kt++) acc2 = wmma_bf16(w2f[kt], bfr[kt], acc2);

      // scatter-add: lane owns edge l15, outputs w*16+8*half .. +7 (contiguous)
      const long long e = tbase + l15;
      if (e < E) {
        float* dbase = agg + (long long)sDst[l15] * H_DIM + w * 16 + 8 * half;
#pragma unroll
        for (int r = 0; r < 8; r++) atomicAdd(dbase + r, acc2[r] + b2v[r]);
      }
      if (threadIdx.x < 16) {
        const long long e2 = tbase + threadIdx.x;
        if (e2 < E) atomicAdd(deg + sDst[threadIdx.x], 1.0f);
      }
    }
    __syncthreads();
  }
}

// ================= update MLP + residual + LayerNorm =================
#define UWAVES 4
#define UTPB (UWAVES * 32)
__global__ void __launch_bounds__(UTPB) upd_kernel(
    const float* __restrict__ node, const float* __restrict__ gst,
    const float* __restrict__ agg, const float* __restrict__ deg,
    const float* __restrict__ ub1, const float* __restrict__ ub2,
    const __bf16* __restrict__ U1t, const __bf16* __restrict__ U2t,
    const float* __restrict__ gam, const float* __restrict__ bet,
    float* __restrict__ out, int N) {
  __shared__ __bf16 ldsA[UWAVES][16 * KU_PAD];  // per-wave padded 16x320 tile
  const int wave = threadIdx.x >> 5;
  const int lane = threadIdx.x & 31;
  const int half = lane >> 4;
  const int l15  = lane & 15;
  const long long tileBase = ((long long)blockIdx.x * UWAVES + wave) * 16;

  __bf16* A = ldsA[wave];
  {  // stage [node | agg/deg | global]
    const int row = lane >> 1, part = lane & 1;
    const long long nd = tileBase + row;
    if (part == 0) {
      unsigned* dp = (unsigned*)(A + row * KU_PAD);
      if (nd < N) {
        const float4* p4 = (const float4*)(node + nd * (long long)H_DIM);
#pragma unroll
        for (int i = 0; i < 32; i++) {
          float4 v = p4[i];
          dp[2 * i] = pk2(v.x, v.y);
          dp[2 * i + 1] = pk2(v.z, v.w);
        }
      } else {
#pragma unroll
        for (int i = 0; i < 64; i++) dp[i] = 0u;
      }
    } else {
      unsigned* dp = (unsigned*)(A + row * KU_PAD + 128);
      if (nd < N) {
        const float inv = 1.0f / fmaxf(deg[nd], 1.0f);
        const float4* p4 = (const float4*)(agg + nd * (long long)H_DIM);
#pragma unroll
        for (int i = 0; i < 32; i++) {
          float4 v = p4[i];
          dp[2 * i] = pk2(v.x * inv, v.y * inv);
          dp[2 * i + 1] = pk2(v.z * inv, v.w * inv);
        }
        const float4* g4 = (const float4*)gst;
#pragma unroll
        for (int i = 0; i < 16; i++) {
          float4 v = g4[i];
          dp[64 + 2 * i] = pk2(v.x, v.y);
          dp[64 + 2 * i + 1] = pk2(v.z, v.w);
        }
      } else {
#pragma unroll
        for (int i = 0; i < 96; i++) dp[i] = 0u;
      }
    }
  }
  __syncthreads();

  const f32x8 z8 = {0.f, 0.f, 0.f, 0.f, 0.f, 0.f, 0.f, 0.f};

  // GEMM1: [16x320] @ [320x128]
  f32x8 acc[8];
#pragma unroll
  for (int nt = 0; nt < 8; nt++) acc[nt] = z8;
  const __bf16* Arow = A + l15 * KU_PAD;
#pragma unroll
  for (int kt = 0; kt < 10; kt++) {
    bf16x16 af = load_a_frag(Arow + kt * 32, half);
#pragma unroll
    for (int nt = 0; nt < 8; nt++) {
      bf16x16 bfr = load_b_frag(U1t + (nt * 16 + l15) * K_UPD + kt * 32, half);
      acc[nt] = wmma_bf16(af, bfr, acc[nt]);
    }
  }

  // bias + ReLU -> hidden (reuse wave region; LDS in-order within wave)
  __bf16* Hd = A;
#pragma unroll
  for (int nt = 0; nt < 8; nt++) {
    const int n = nt * 16 + l15;
    const float bb = ub1[n];
#pragma unroll
    for (int r = 0; r < 8; r++) {
      float v = fmaxf(acc[nt][r] + bb, 0.f);
      Hd[(r + 8 * half) * H_PAD + n] = tobf(v);
    }
  }

  // GEMM2: [16x128] @ [128x128]
  f32x8 acc2[8];
#pragma unroll
  for (int nt = 0; nt < 8; nt++) acc2[nt] = z8;
  const __bf16* Hrow = Hd + l15 * H_PAD;
#pragma unroll
  for (int kt = 0; kt < 4; kt++) {
    bf16x16 af = load_a_frag(Hrow + kt * 32, half);
#pragma unroll
    for (int nt = 0; nt < 8; nt++) {
      bf16x16 bfr = load_b_frag(U2t + (nt * 16 + l15) * H_DIM + kt * 32, half);
      acc2[nt] = wmma_bf16(af, bfr, acc2[nt]);
    }
  }

  // residual, then LayerNorm per row via half-wave shuffle reduction
  float xv[8][8];
#pragma unroll
  for (int nt = 0; nt < 8; nt++) {
    const int n = nt * 16 + l15;
    const float bb = ub2[n];
#pragma unroll
    for (int r = 0; r < 8; r++) {
      const long long nd = tileBase + r + 8 * half;
      float x = acc2[nt][r] + bb;
      x += (nd < N) ? node[nd * (long long)H_DIM + n] : 0.f;
      xv[nt][r] = x;
    }
  }
#pragma unroll
  for (int r = 0; r < 8; r++) {
    float s = 0.f, s2 = 0.f;
#pragma unroll
    for (int nt = 0; nt < 8; nt++) {
      float x = xv[nt][r];
      s += x;
      s2 += x * x;
    }
#pragma unroll
    for (int mask = 1; mask < 16; mask <<= 1) {  // stays within 16-lane half
      s  += __shfl_xor(s, mask, 32);
      s2 += __shfl_xor(s2, mask, 32);
    }
    const float mu  = s * (1.0f / H_DIM);
    const float var = s2 * (1.0f / H_DIM) - mu * mu;
    const float rs  = rsqrtf(var + 1e-5f);
    const long long nd = tileBase + r + 8 * half;
    if (nd < N) {
#pragma unroll
      for (int nt = 0; nt < 8; nt++) {
        const int n = nt * 16 + l15;
        out[nd * (long long)H_DIM + n] = (xv[nt][r] - mu) * rs * gam[n] + bet[n];
      }
    }
  }
}

extern "C" void kernel_launch(void* const* d_in, const int* in_sizes, int n_in,
                              void* d_out, int out_size, void* d_ws, size_t ws_size,
                              hipStream_t stream) {
  (void)n_in; (void)out_size; (void)ws_size;
  const float*     node = (const float*)d_in[0];
  const long long* eidx = (const long long*)d_in[1];
  const float*     est  = (const float*)d_in[2];
  const float*     gst  = (const float*)d_in[3];
  const float*     mw1  = (const float*)d_in[4];
  const float*     mb1  = (const float*)d_in[5];
  const float*     mw2  = (const float*)d_in[6];
  const float*     mb2  = (const float*)d_in[7];
  const float*     uw1  = (const float*)d_in[8];
  const float*     ub1  = (const float*)d_in[9];
  const float*     uw2  = (const float*)d_in[10];
  const float*     ub2  = (const float*)d_in[11];
  const float*     gam  = (const float*)d_in[12];
  const float*     bet  = (const float*)d_in[13];

  const int N = in_sizes[0] / H_DIM;
  const int E = in_sizes[2] / 64;

  char* ws = (char*)d_ws;
  auto al = [](size_t x) { return (x + 255) & ~(size_t)255; };
  size_t off = 0;
  float* agg = (float*)(ws + off); off += al((size_t)N * H_DIM * sizeof(float));
  float* deg = (float*)(ws + off); off += al((size_t)N * sizeof(float));
  __bf16* W1t = (__bf16*)(ws + off); off += al((size_t)K_MSG * H_DIM * 2);
  __bf16* W2t = (__bf16*)(ws + off); off += al((size_t)H_DIM * H_DIM * 2);
  __bf16* U1t = (__bf16*)(ws + off); off += al((size_t)K_UPD * H_DIM * 2);
  __bf16* U2t = (__bf16*)(ws + off); off += al((size_t)H_DIM * H_DIM * 2);

  hipMemsetAsync(agg, 0, (size_t)N * H_DIM * sizeof(float), stream);
  hipMemsetAsync(deg, 0, (size_t)N * sizeof(float), stream);

  prep_kernel<<<(K_MSG * H_DIM + 255) / 256, 256, 0, stream>>>(mw1, W1t, K_MSG, H_DIM);
  prep_kernel<<<(H_DIM * H_DIM + 255) / 256, 256, 0, stream>>>(mw2, W2t, H_DIM, H_DIM);
  prep_kernel<<<(K_UPD * H_DIM + 255) / 256, 256, 0, stream>>>(uw1, U1t, K_UPD, H_DIM);
  prep_kernel<<<(H_DIM * H_DIM + 255) / 256, 256, 0, stream>>>(uw2, U2t, H_DIM, H_DIM);

  const long long etiles = ((long long)E + 15) / 16;
  const int eblocks = (int)(etiles < 2048 ? etiles : 2048);
  msg_kernel<<<eblocks, 256, 0, stream>>>(node, eidx, est, gst, mb1, mb2, W1t, W2t,
                                          agg, deg, E);

  const int ntiles  = (N + 15) / 16;
  const int nblocks = (ntiles + UWAVES - 1) / UWAVES;
  upd_kernel<<<nblocks, UTPB, 0, stream>>>(node, gst, agg, deg, ub1, ub2, U1t, U2t,
                                           gam, bet, (float*)d_out, N);
}
